// DifferentialBandAttention_2894807958198
// MI455X (gfx1250) — compile-verified, matrix operand order checked by reading, not run on hardware
//
#include <hip/hip_runtime.h>
#include <hip/hip_bf16.h>

typedef _Float16 f16;
typedef __attribute__((ext_vector_type(16))) _Float16 v16h;
typedef __attribute__((ext_vector_type(8)))  float    v8f;
typedef unsigned int v4u __attribute__((ext_vector_type(4)));
typedef int          v8i __attribute__((ext_vector_type(8)));
typedef int          v4i __attribute__((ext_vector_type(4)));

#define B_ 4
#define N_ 128
#define T_ 512
#define K_ 30
#define SCALE_INV 0.25f  // 1/sqrt(HEAD_DIM=16)

#define P_STRIDE  520    // 512 + 8 f32  (bank decorrelation)
#define VT_STRIDE 528    // 512 + 16 f16 (32B-aligned rows, bank decorrelation)
#define QS_STRIDE 144    // 128 + 16 f16 (matches TDM pad: 8 dwords per 64-dword row)

// ---------------- workspace layout (bytes) ----------------
constexpr size_t SZ_XH  = (size_t)K_ * B_ * T_ * N_ * 2;
constexpr size_t SZ_WQH = (size_t)K_ * N_ * N_ * 2;
constexpr size_t SZ_WVH = (size_t)K_ * 64 * N_ * 2;
constexpr size_t SZ_QH  = SZ_XH;
constexpr size_t SZ_VH  = (size_t)K_ * B_ * T_ * 64 * 2;
constexpr size_t SZ_O   = (size_t)K_ * B_ * T_ * 64 * 4;

constexpr size_t OFF_XH  = 0;
constexpr size_t OFF_WQH = OFF_XH  + SZ_XH;
constexpr size_t OFF_WKH = OFF_WQH + SZ_WQH;
constexpr size_t OFF_WVH = OFF_WKH + SZ_WQH;
constexpr size_t OFF_WOH = OFF_WVH + SZ_WVH;
constexpr size_t OFF_QH  = OFF_WOH + SZ_WVH;
constexpr size_t OFF_KH  = OFF_QH  + SZ_QH;
constexpr size_t OFF_VH  = OFF_KH  + SZ_QH;
constexpr size_t OFF_O1  = OFF_VH  + SZ_VH;
constexpr size_t OFF_O2  = OFF_O1  + SZ_O;
constexpr size_t OFF_ST  = OFF_O2  + SZ_O;
constexpr size_t OFF_LAM = OFF_ST  + (size_t)K_ * B_ * 8 * 4;

// ---------------- WMMA helpers ----------------
__device__ inline v8f wmma_f16(v16h a, v16h b, v8f c) {
  return __builtin_amdgcn_wmma_f32_16x16x32_f16(
      false, a, false, b, (short)0, c, false, false);
}

// Fragment where the reduction axis is contiguous: lane L -> row row0+L%16,
// 16 contiguous f16 at col0 + (L/16)*16.  (works for global and LDS pointers)
__device__ inline v16h load_frag_row(const f16* base, int stride, int row0, int col0, int lane) {
  const f16* p = base + (size_t)(row0 + (lane & 15)) * stride + col0 + ((lane >> 4) << 4);
  return *(const v16h*)p;
}

// A-fragment from an f32 LDS tile (convert on the fly).
__device__ inline v16h load_frag_lds_f32(const float* base, int stride, int row0, int col0, int lane) {
  const float* p = base + (size_t)(row0 + (lane & 15)) * stride + col0 + ((lane >> 4) << 4);
  v16h r;
#pragma unroll
  for (int i = 0; i < 16; ++i) r[i] = (f16)p[i];
  return r;
}

// C/D tile: VGPR j -> row row0 + (lane/16)*8 + j, col col0 + lane%16
__device__ inline void store_tile_f16(f16* base, int stride, int row0, int col0, int lane, v8f c) {
  int col = col0 + (lane & 15);
  int r0  = row0 + ((lane >> 4) << 3);
#pragma unroll
  for (int j = 0; j < 8; ++j) base[(size_t)(r0 + j) * stride + col] = (f16)c[j];
}

__device__ inline void store_tile_f32(float* base, int stride, int row0, int col0, int lane, v8f c) {
  int col = col0 + (lane & 15);
  int r0  = row0 + ((lane >> 4) << 3);
#pragma unroll
  for (int j = 0; j < 8; ++j) base[(size_t)(r0 + j) * stride + col] = c[j];
}

// ---- TDM: 2D tile load global->LDS, data_size=4B, optional per-row LDS pad ----
// rows x cols given in 4-byte units; row_stride in 4-byte units.
__device__ inline void tdm_load_2d(unsigned lds_byte_off, const void* gptr,
                                   unsigned cols4, unsigned rows,
                                   unsigned row_stride4, bool pad_8dw_per_64dw) {
  unsigned long long ga = (unsigned long long)(size_t)gptr;
  v4u g0 = { 1u,                                  // count=1, user mode
             lds_byte_off,                        // lds_addr
             (unsigned)(ga & 0xffffffffu),        // global_addr[31:0]
             (unsigned)((ga >> 32) & 0x01ffffffu) | (2u << 30) };  // addr[56:32] | type=2
  unsigned d0 = (2u << 16);                       // data_size = 4 bytes
  if (pad_8dw_per_64dw) d0 |= (1u << 20) | (5u << 22) | (7u << 25); // pad_en, every 64dw, +8dw
  v8i g1 = { (int)d0,
             (int)((cols4 & 0xffffu) << 16),                       // tensor_dim0[15:0]
             (int)((cols4 >> 16) | ((rows & 0xffffu) << 16)),      // dim0 hi | tensor_dim1 lo
             (int)((rows >> 16) | ((cols4 & 0xffffu) << 16)),      // dim1 hi | tile_dim0
             (int)(rows & 0xffffu),                                // tile_dim1 (tile_dim2=0)
             (int)row_stride4,                                     // tensor_dim0_stride lo
             0, 0 };
  v4i z4 = { 0, 0, 0, 0 };
#if __clang_major__ >= 23
  v8i z8 = { 0, 0, 0, 0, 0, 0, 0, 0 };
  __builtin_amdgcn_tensor_load_to_lds(g0, g1, z4, z4, z8, 0);
#else
  __builtin_amdgcn_tensor_load_to_lds(g0, g1, z4, z4, 0);
#endif
}

// ---------------- kernel 0a: x [B,N,T,K] f32 -> xh [K,B,T,N] f16 ----------------
__global__ void prep_x_kernel(const float* __restrict__ x, f16* __restrict__ xh) {
  int idx = blockIdx.x * blockDim.x + threadIdx.x;
  if (idx >= K_ * B_ * T_ * N_) return;
  int n    = idx & (N_ - 1);
  int t    = (idx >> 7) & (T_ - 1);
  int rest = idx >> 16;
  int b    = rest & 3;
  int k    = rest >> 2;
  xh[idx] = (f16)x[(((size_t)b * N_ + n) * T_ + t) * K_ + k];
}

// ---------------- kernel 0b: weights f32 -> f16 ----------------
__global__ void prep_w_kernel(const float* __restrict__ Wq, const float* __restrict__ Wk,
                              const float* __restrict__ Wv, const float* __restrict__ Wo,
                              f16* __restrict__ wqh, f16* __restrict__ wkh,
                              f16* __restrict__ wvh, f16* __restrict__ woh) {
  int idx = blockIdx.x * blockDim.x + threadIdx.x;
  if (idx < K_ * N_ * N_) { wqh[idx] = (f16)Wq[idx]; wkh[idx] = (f16)Wk[idx]; }
  if (idx < K_ * 64 * N_) { wvh[idx] = (f16)Wv[idx]; woh[idx] = (f16)Wo[idx]; }
}

// ---------------- kernel 0c: init stats accumulators ----------------
__global__ void init_stats_kernel(float* __restrict__ stats, float* __restrict__ lam) {
  int i = threadIdx.x;
  if (i < K_ * B_) {
    float* s = stats + i * 8;
    s[0] = 0.f; s[1] = 0.f; s[2] = -__builtin_inff(); s[3] = __builtin_inff();
    s[4] = 0.f; s[5] = 0.f; s[6] = -__builtin_inff(); s[7] = __builtin_inff();
    lam[i] = 0.f;
  }
}

// ---------------- kernel 1: q/k/v projections (WMMA) ----------------
__global__ __launch_bounds__(256) void proj_kernel(
    const f16* __restrict__ xh, const f16* __restrict__ wqh, const f16* __restrict__ wkh,
    const f16* __restrict__ wvh, f16* __restrict__ qh, f16* __restrict__ kh,
    f16* __restrict__ vh) {
  const int lane = threadIdx.x & 31;
  const int wave = threadIdx.x >> 5;
  const int kb   = blockIdx.y;
  const int t0   = blockIdx.x * 32;
  const int kband = kb >> 2;

  const f16* xb = xh + (size_t)kb * T_ * N_;
  const f16* WQ = wqh + (size_t)kband * N_ * N_;
  const f16* WK = wkh + (size_t)kband * N_ * N_;
  const f16* WV = wvh + (size_t)kband * 64 * N_;
  f16* q  = qh + (size_t)kb * T_ * N_;
  f16* kk = kh + (size_t)kb * T_ * N_;
  f16* v  = vh + (size_t)kb * T_ * 64;

  for (int i = 0; i < 5; ++i) {
    int tile  = wave + i * 8;
    int which = tile >> 4;                 // 0=q, 1=k, 2=v
    int local = (which < 2) ? (tile & 15) : (tile - 32);
    int nt = local >> 1, mt = local & 1;
    const f16* W = (which == 0) ? WQ : (which == 1) ? WK : WV;
    v8f acc = {};
#pragma unroll
    for (int kc = 0; kc < N_; kc += 32) {
      v16h a = load_frag_row(xb, N_, t0 + 16 * mt, kc, lane);
      v16h b = load_frag_row(W,  N_, 16 * nt,      kc, lane);
      acc = wmma_f16(a, b, acc);
    }
    if (which == 0)      store_tile_f16(q,  N_, t0 + 16 * mt, 16 * nt, lane, acc);
    else if (which == 1) store_tile_f16(kk, N_, t0 + 16 * mt, 16 * nt, lane, acc);
    else                 store_tile_f16(v,  64, t0 + 16 * mt, 16 * nt, lane, acc);
  }
}

// ---------------- kernel 2: attention ----------------
__global__ __launch_bounds__(256) void attn_kernel(
    const f16* __restrict__ qh, const f16* __restrict__ kh, const f16* __restrict__ vh,
    float* __restrict__ o1, float* __restrict__ o2, float* __restrict__ stats) {
  __shared__ __align__(128) float P[32 * P_STRIDE];     // probability tile
  __shared__ __align__(128) f16  Vt[64 * VT_STRIDE];    // V transposed: Vt[n][s]
  __shared__ __align__(128) f16  Qs[32 * QS_STRIDE];    // Q rows (TDM-staged, padded)
  __shared__ float redA[256], redB[256];

  const int lane = threadIdx.x & 31;
  const int wave = threadIdx.x >> 5;
  const int kb   = blockIdx.y;
  const int t0   = blockIdx.x * 32;

  const f16* q    = qh + (size_t)kb * T_ * N_;
  const f16* kmat = kh + (size_t)kb * T_ * N_;
  const f16* v    = vh + (size_t)kb * T_ * 64;

  // ---- TDM: stage Q rows [32 x 128 f16] into LDS with +8dw/row padding ----
  if (threadIdx.x < 32) {
    unsigned lds_off = (unsigned)(size_t)&Qs[0];   // low 32 bits of generic = LDS byte offset
    tdm_load_2d(lds_off, q + (size_t)t0 * N_,
                /*cols4=*/N_ / 2, /*rows=*/32, /*row_stride4=*/N_ / 2,
                /*pad=*/true);
    __builtin_amdgcn_s_wait_tensorcnt(0);
  }
  // ---- cooperative V transpose into LDS (overlaps with TDM) ----
  for (int e = threadIdx.x; e < 64 * 512; e += 256) {
    int n = e & 63, s = e >> 6;                    // coalesced global reads
    Vt[n * VT_STRIDE + s] = v[(size_t)s * 64 + n];
  }
  __syncthreads();

  for (int pass = 0; pass < 2; ++pass) {
    const int hoff = pass * 64;
    float* o  = (pass == 0 ? o1 : o2) + (size_t)kb * T_ * 64;
    float* st = stats + kb * 8 + pass * 4;

    // ---- logits: S[32,512] = Q[32,64] * K^T ; 64 tiles, 8 per wave ----
    for (int i = 0; i < 8; ++i) {
      int tile = wave + i * 8;
      int mt = tile & 1, stile = tile >> 1;
      v8f acc = {};
#pragma unroll
      for (int kc = 0; kc < 64; kc += 32) {
        v16h a = load_frag_row((const f16*)Qs, QS_STRIDE, 16 * mt, hoff + kc, lane);
        v16h b = load_frag_row(kmat, N_, 16 * stile, hoff + kc, lane);
        acc = wmma_f16(a, b, acc);
      }
      int col = 16 * stile + (lane & 15);
      int r0  = 16 * mt + ((lane >> 4) << 3);
#pragma unroll
      for (int j = 0; j < 8; ++j) P[(r0 + j) * P_STRIDE + col] = acc[j] * SCALE_INV;
    }
    __syncthreads();

    // ---- row softmax + stats: 8 threads per row ----
    {
      int row = threadIdx.x >> 3;       // 0..31
      int seg = threadIdx.x & 7;        // 0..7 -> cols [seg*64, seg*64+64)
      float* prow = P + row * P_STRIDE;
      int c0 = seg * 64;

      float m = -__builtin_inff();
      for (int s = c0; s < c0 + 64; ++s) m = fmaxf(m, prow[s]);
      redA[threadIdx.x] = m;
      __syncthreads();
      float mx = redA[row * 8];
#pragma unroll
      for (int j = 1; j < 8; ++j) mx = fmaxf(mx, redA[row * 8 + j]);

      float sloc = 0.f;
      for (int s = c0; s < c0 + 64; ++s) {
        float e = __expf(prow[s] - mx); prow[s] = e; sloc += e;
      }
      redB[threadIdx.x] = sloc;
      __syncthreads();
      float tot = redB[row * 8];
#pragma unroll
      for (int j = 1; j < 8; ++j) tot += redB[row * 8 + j];
      float inv = 1.0f / tot;

      float rsum = 0.f, rsq = 0.f, rmx = -__builtin_inff(), rmn = __builtin_inff();
      for (int s = c0; s < c0 + 64; ++s) {
        float p0 = prow[s] * inv; prow[s] = p0;
        rsum += p0; rsq += p0 * p0; rmx = fmaxf(rmx, p0); rmn = fminf(rmn, p0);
      }
      atomicAdd(&st[0], rsum);
      atomicAdd(&st[1], rsq);
      atomicMax((int*)&st[2], __float_as_int(rmx));   // softmax outputs > 0
      atomicMin((int*)&st[3], __float_as_int(rmn));
    }
    __syncthreads();

    // ---- O[32,64] = P[32,512] * V[512,64] ; 8 tiles, 1 per wave ----
    {
      int mt = wave & 1, nt = wave >> 1;
      v8f acc = {};
      for (int sc = 0; sc < 512; sc += 32) {
        v16h a = load_frag_lds_f32(P, P_STRIDE, 16 * mt, sc, lane);
        v16h b = load_frag_row((const f16*)Vt, VT_STRIDE, 16 * nt, sc, lane); // B[s][n]=Vt[n][s]
        acc = wmma_f16(a, b, acc);
      }
      store_tile_f32(o, 64, t0 + 16 * mt, 16 * nt, lane, acc);
    }
    __syncthreads();   // P/red reused by next pass
  }
}

// ---------------- kernel 3: stats -> MLP -> lambda ----------------
__global__ void lam_kernel(const float* __restrict__ stats,
                           const float* __restrict__ W1, const float* __restrict__ b1,
                           const float* __restrict__ W2, const float* __restrict__ b2,
                           float* __restrict__ lam) {
  int i = blockIdx.x * blockDim.x + threadIdx.x;
  if (i >= K_ * B_) return;
  int kband = i >> 2;
  const float* s = stats + i * 8;
  const float n = (float)T_ * (float)T_;
  float feat[8];
#pragma unroll
  for (int p = 0; p < 2; ++p) {
    float sum = s[p * 4 + 0], sq = s[p * 4 + 1];
    float mean = sum / n;
    float var  = (sq - sum * sum / n) / (n - 1.0f);
    feat[p * 4 + 0] = mean;
    feat[p * 4 + 1] = sqrtf(fmaxf(var, 0.f));
    feat[p * 4 + 2] = s[p * 4 + 2];
    feat[p * 4 + 3] = s[p * 4 + 3];
  }
  const float* w1  = W1 + kband * 16 * 8;
  const float* bb1 = b1 + kband * 16;
  const float* w2  = W2 + kband * 16;
  float z = b2[kband];
  for (int j = 0; j < 16; ++j) {
    float h = bb1[j];
#pragma unroll
    for (int qq = 0; qq < 8; ++qq) h += w1[j * 8 + qq] * feat[qq];
    z += w2[j] * fmaxf(h, 0.f);
  }
  lam[i] = 1.0f / (1.0f + __expf(-z));
}

// ---------------- kernel 4: out = (o1 - lam*o2) @ Wo^T + bo ----------------
__global__ __launch_bounds__(256) void out_kernel(
    const float* __restrict__ o1, const float* __restrict__ o2,
    const f16* __restrict__ woh, const float* __restrict__ bo,
    const float* __restrict__ lam, float* __restrict__ out) {
  __shared__ __align__(128) f16 D[32 * 64];
  const int lane = threadIdx.x & 31;
  const int wave = threadIdx.x >> 5;
  const int kb   = blockIdx.y;
  const int t0   = blockIdx.x * 32;
  const int kband = kb >> 2, bb = kb & 3;

  const float l = lam[kb];
  const float* p1 = o1 + (size_t)kb * T_ * 64 + (size_t)t0 * 64;
  const float* p2 = o2 + (size_t)kb * T_ * 64 + (size_t)t0 * 64;
  for (int i = threadIdx.x; i < 32 * 64; i += 256)
    D[i] = (f16)(p1[i] - l * p2[i]);
  __syncthreads();

  const f16* WO = woh + (size_t)kband * N_ * 64;
  const float* bvec = bo + kband * N_;

  for (int i = 0; i < 2; ++i) {
    int tile = wave + i * 8;
    int mt = tile & 1, nt = tile >> 1;
    v8f acc = {};
#pragma unroll
    for (int kc = 0; kc < 64; kc += 32) {
      v16h a = load_frag_row((const f16*)D, 64, 16 * mt, kc, lane);
      v16h b = load_frag_row(WO, 64, 16 * nt, kc, lane);
      acc = wmma_f16(a, b, acc);
    }
    int m     = 16 * nt + (lane & 15);
    int trow0 = t0 + 16 * mt + ((lane >> 4) << 3);
    float bias = bvec[m];
#pragma unroll
    for (int j = 0; j < 8; ++j) {
      int t = trow0 + j;
      out[(((size_t)bb * N_ + m) * T_ + t) * K_ + kband] = acc[j] + bias;
    }
  }
}

// ---------------- host side ----------------
extern "C" void kernel_launch(void* const* d_in, const int* in_sizes, int n_in,
                              void* d_out, int out_size, void* d_ws, size_t ws_size,
                              hipStream_t stream) {
  (void)in_sizes; (void)n_in; (void)out_size; (void)ws_size;
  const float* x  = (const float*)d_in[0];
  const float* Wq = (const float*)d_in[1];
  const float* Wk = (const float*)d_in[2];
  const float* Wv = (const float*)d_in[3];
  const float* Wo = (const float*)d_in[4];
  const float* bo = (const float*)d_in[5];
  const float* W1 = (const float*)d_in[6];
  const float* b1 = (const float*)d_in[7];
  const float* W2 = (const float*)d_in[8];
  const float* b2 = (const float*)d_in[9];

  char* ws = (char*)d_ws;
  f16*   xh    = (f16*)  (ws + OFF_XH);
  f16*   wqh   = (f16*)  (ws + OFF_WQH);
  f16*   wkh   = (f16*)  (ws + OFF_WKH);
  f16*   wvh   = (f16*)  (ws + OFF_WVH);
  f16*   woh   = (f16*)  (ws + OFF_WOH);
  f16*   qh    = (f16*)  (ws + OFF_QH);
  f16*   kh    = (f16*)  (ws + OFF_KH);
  f16*   vh    = (f16*)  (ws + OFF_VH);
  float* o1    = (float*)(ws + OFF_O1);
  float* o2    = (float*)(ws + OFF_O2);
  float* stats = (float*)(ws + OFF_ST);
  float* lamb  = (float*)(ws + OFF_LAM);

  prep_x_kernel<<<dim3((K_ * B_ * T_ * N_ + 255) / 256), dim3(256), 0, stream>>>(x, xh);
  prep_w_kernel<<<dim3((K_ * N_ * N_ + 255) / 256), dim3(256), 0, stream>>>(
      Wq, Wk, Wv, Wo, wqh, wkh, wvh, woh);
  init_stats_kernel<<<dim3(1), dim3(256), 0, stream>>>(stats, lamb);

  proj_kernel<<<dim3(T_ / 32, K_ * B_), dim3(256), 0, stream>>>(
      xh, wqh, wkh, wvh, qh, kh, vh);
  attn_kernel<<<dim3(T_ / 32, K_ * B_), dim3(256), 0, stream>>>(
      qh, kh, vh, o1, o2, stats);
  lam_kernel<<<dim3(1), dim3(128), 0, stream>>>(stats, W1, b1, W2, b2, lamb);
  out_kernel<<<dim3(T_ / 32, K_ * B_), dim3(256), 0, stream>>>(
      o1, o2, woh, bo, lamb, (float*)d_out);
}
